// MCxM_GNN_17944373363255
// MI455X (gfx1250) — matrix operand order repaired and checked by reading, not compiled, not run on hardware
//
#include <hip/hip_runtime.h>

#define N_NODES   50000
#define N_EDGES   1600000
#define DIM       128
#define N_GRAPHS  64
#define N_LAYERS  3
#define BN_EPS    1e-5f

typedef __attribute__((ext_vector_type(16))) __bf16 v16bf;
typedef __attribute__((ext_vector_type(8)))  __bf16 v8bf;
typedef __attribute__((ext_vector_type(8)))  float  v8f;

// ---------------- utility ----------------
__global__ void k_zero(float* __restrict__ p, int n) {
    int i = blockIdx.x * blockDim.x + threadIdx.x;
    if (i < n) p[i] = 0.0f;
}

// ---------------- degree ----------------
__global__ void k_deg(const int* __restrict__ dst, float* __restrict__ deg, int E) {
    int i = blockIdx.x * blockDim.x + threadIdx.x;
    if (i < E) atomicAdd(&deg[dst[i]], 1.0f);
}

__global__ void k_disqrt(float* __restrict__ deg, int n) {
    int i = blockIdx.x * blockDim.x + threadIdx.x;
    if (i < n) deg[i] = rsqrtf(deg[i] + 1.0f);   // self-loop: deg+1
}

// ---------------- masked GEMM: hw = (h*mask) @ W, bf16 WMMA ----------------
// Block = 256 threads = 8 waves. Wave w owns N-tile w (16 output columns);
// its 4 B fragments stay in registers for the whole kernel. The 16x128 A tile
// (mask applied, converted to bf16) is staged ONCE per block into a
// double-buffered LDS tile; each wave pulls A fragments via ds_load_b128.
// Two independent accumulator chains interleave the WMMAs to avoid D->C
// serialization stalls.
__global__ __launch_bounds__(256) void k_gemm_bf16(
    const float* __restrict__ h, const float* __restrict__ mask,
    const float* __restrict__ W, float* __restrict__ hw, int nrows) {
    __shared__ __bf16 Wl[DIM * DIM];        // 32 KB: W in bf16
    __shared__ __bf16 At[2][16 * DIM];      // 2 x 4 KB: double-buffered A tile

    // stage W -> LDS (bf16)
    for (int i = threadIdx.x; i < DIM * DIM; i += 256) Wl[i] = (__bf16)W[i];

    const int lane = threadIdx.x & 31;
    const int nt   = threadIdx.x >> 5;   // wave id == N-tile
    const int lo   = lane & 15;
    const int hi   = lane >> 4;          // 0 or 1

    const int ntiles = nrows >> 4;       // 50000/16 = 3125, exact

    // cooperative A-tile stage: 256 threads x 8 floats = 16x128 tile
    auto stage = [&](int t, int buf) {
        const int idx = threadIdx.x << 3;        // 0..2040
        const int row = idx >> 7;                // 0..15
        const int col = idx & 127;               // multiple of 8
        const int m   = (t << 4) + row;
        const float mk = mask[m];
        const float* hp = h + (size_t)m * DIM + col;
        const float4 f0 = *(const float4*)(hp);
        const float4 f1 = *(const float4*)(hp + 4);
        v8bf o;
        o[0] = (__bf16)(f0.x * mk); o[1] = (__bf16)(f0.y * mk);
        o[2] = (__bf16)(f0.z * mk); o[3] = (__bf16)(f0.w * mk);
        o[4] = (__bf16)(f1.x * mk); o[5] = (__bf16)(f1.y * mk);
        o[6] = (__bf16)(f1.z * mk); o[7] = (__bf16)(f1.w * mk);
        *(v8bf*)&At[buf][row * DIM + col] = o;
    };

    int t = blockIdx.x;
    if (t < ntiles) stage(t, 0);
    __syncthreads();                      // W + first A tile visible

    // B fragment: lane (0-15) holds column n = nt*16+lo, k = kb*32 + hi*16 + j
    v16bf bfrag[4];
#pragma unroll
    for (int kb = 0; kb < 4; ++kb) {
        const int kbase = kb * 32 + hi * 16;
#pragma unroll
        for (int j = 0; j < 16; ++j)
            bfrag[kb][j] = Wl[(kbase + j) * DIM + nt * 16 + lo];
    }

    int p = 0;
    while (t < ntiles) {
        const int tn = t + gridDim.x;
        const int m0 = t << 4;

        v8f acc0 = {0.f, 0.f, 0.f, 0.f, 0.f, 0.f, 0.f, 0.f};
        v8f acc1 = {0.f, 0.f, 0.f, 0.f, 0.f, 0.f, 0.f, 0.f};
#pragma unroll
        for (int kb = 0; kb < 4; ++kb) {
            // A frag (contiguous K, matching B): elems 0..15 -> K=k0..k0+15,
            // row M = lo, k0 = kb*32 + hi*16
            const int k0 = kb * 32 + hi * 16;
            const v8bf a_lo = *(const v8bf*)&At[p][lo * DIM + k0];
            const v8bf a_hi = *(const v8bf*)&At[p][lo * DIM + k0 + 8];
            const v16bf a = __builtin_shufflevector(
                a_lo, a_hi, 0, 1, 2, 3, 4, 5, 6, 7, 8, 9, 10, 11, 12, 13, 14, 15);
            if (kb & 1)
                acc1 = __builtin_amdgcn_wmma_f32_16x16x32_bf16(
                    false, a, false, bfrag[kb], (short)0, acc1, false, false);
            else
                acc0 = __builtin_amdgcn_wmma_f32_16x16x32_bf16(
                    false, a, false, bfrag[kb], (short)0, acc0, false, false);
        }

        if (tn < ntiles) stage(tn, p ^ 1);   // overlap next-tile staging

        acc0 = acc0 + acc1;
        // D layout: lane -> col nt*16+lo, rows m0 + r + 8*hi
        float* op = hw + (size_t)(m0 + hi * 8) * DIM + nt * 16 + lo;
#pragma unroll
        for (int r = 0; r < 8; ++r) op[(size_t)r * DIM] = acc0[r];

        __syncthreads();                     // readers of p done, buffer p^1 ready
        t = tn;
        p ^= 1;
    }
}

// ---------------- edge scatter: agg[dst] += hw[src] * dis[src]*dis[dst] ----------------
__global__ void k_scatter(const int* __restrict__ src, const int* __restrict__ dst,
                          const float* __restrict__ dis, const float* __restrict__ hw,
                          float* __restrict__ agg, int E) {
    long long t = (long long)blockIdx.x * blockDim.x + threadIdx.x;
    int e = (int)(t >> 5);
    if (e >= E) return;
    int c0 = ((int)t & 31) * 4;
    int s = src[e], d = dst[e];
    float w = dis[s] * dis[d];
    const float4 v = *(const float4*)(hw + (size_t)s * DIM + c0);
    float* ap = agg + (size_t)d * DIM + c0;
    atomicAdd(ap + 0, v.x * w);
    atomicAdd(ap + 1, v.y * w);
    atomicAdd(ap + 2, v.z * w);
    atomicAdd(ap + 3, v.w * w);
}

// ---------------- self-loop + bias, fused BN statistics ----------------
__global__ __launch_bounds__(128) void k_selfbias_stats(
    float* __restrict__ agg, const float* __restrict__ hw,
    const float* __restrict__ dis, const float* __restrict__ b,
    float* __restrict__ bnsum, float* __restrict__ bnsq, int nrows) {
    const int c = threadIdx.x;           // channel 0..127
    const float bias = b[c];
    float s = 0.f, s2 = 0.f;
    for (int n = blockIdx.x; n < nrows; n += gridDim.x) {
        float d = dis[n];
        size_t idx = (size_t)n * DIM + c;
        float v = agg[idx] + hw[idx] * d * d + bias;
        agg[idx] = v;
        s += v; s2 += v * v;
    }
    atomicAdd(&bnsum[c], s);
    atomicAdd(&bnsq[c],  s2);
}

__global__ void k_bn_finalize(const float* __restrict__ bnsum, const float* __restrict__ bnsq,
                              const float* __restrict__ gamma, const float* __restrict__ beta,
                              float* __restrict__ bna, float* __restrict__ bnb, int nrows) {
    int c = threadIdx.x;
    if (c < DIM) {
        float inv = 1.0f / (float)nrows;
        float mu  = bnsum[c] * inv;
        float var = bnsq[c] * inv - mu * mu;
        float sc  = gamma[c] * rsqrtf(var + BN_EPS);
        bna[c] = sc;
        bnb[c] = beta[c] - mu * sc;
    }
}

__global__ void k_bn_relu(const float* __restrict__ agg, const float* __restrict__ bna,
                          const float* __restrict__ bnb, float* __restrict__ hout, int total) {
    int i = blockIdx.x * blockDim.x + threadIdx.x;
    if (i < total) {
        int c = i & (DIM - 1);
        hout[i] = fmaxf(fmaf(agg[i], bna[c], bnb[c]), 0.0f);
    }
}

// ---------------- pooling + readout ----------------
__global__ __launch_bounds__(128) void k_pool(const float* __restrict__ h,
                                              const float* __restrict__ mask,
                                              const int* __restrict__ batch,
                                              float* __restrict__ pooled, int nrows) {
    const int c = threadIdx.x;
    for (int n = blockIdx.x; n < nrows; n += gridDim.x) {
        float v = h[(size_t)n * DIM + c] * mask[n];
        atomicAdd(&pooled[(size_t)batch[n] * DIM + c], v);
    }
}

__global__ void k_counts(const int* __restrict__ batch, float* __restrict__ counts, int nrows) {
    int n = blockIdx.x * blockDim.x + threadIdx.x;
    if (n < nrows) atomicAdd(&counts[batch[n]], 1.0f);
}

__global__ void k_out(const float* __restrict__ pooled, const float* __restrict__ counts,
                      const float* __restrict__ Wout, const float* __restrict__ bout,
                      float* __restrict__ out) {
    int g = threadIdx.x;
    if (g < N_GRAPHS) {
        float s = 0.f;
        for (int c = 0; c < DIM; ++c) s += pooled[(size_t)g * DIM + c] * Wout[c];
        out[g] = s / fmaxf(counts[g], 1.0f) + bout[0];
    }
}

// ---------------- launcher ----------------
extern "C" void kernel_launch(void* const* d_in, const int* in_sizes, int n_in,
                              void* d_out, int out_size, void* d_ws, size_t ws_size,
                              hipStream_t stream) {
    (void)in_sizes; (void)n_in; (void)out_size; (void)ws_size;
    const float* x     = (const float*)d_in[0];
    const int*   ei    = (const int*)d_in[1];
    const int*   src   = ei;
    const int*   dst   = ei + N_EDGES;
    const float* mask  = (const float*)d_in[2];
    const int*   batch = (const int*)d_in[3];
    const float* W     = (const float*)d_in[4];
    const float* b     = (const float*)d_in[5];
    const float* gamma = (const float*)d_in[6];
    const float* beta  = (const float*)d_in[7];
    const float* Wout  = (const float*)d_in[8];
    const float* bout  = (const float*)d_in[9];
    float*       out   = (float*)d_out;

    float* ws     = (float*)d_ws;
    float* dis    = ws;                                     // N
    float* hbuf   = dis  + N_NODES;                         // N*D
    float* hwbuf  = hbuf + (size_t)N_NODES * DIM;           // N*D
    float* aggbuf = hwbuf + (size_t)N_NODES * DIM;          // N*D
    float* bnsum  = aggbuf + (size_t)N_NODES * DIM;         // D
    float* bnsq   = bnsum + DIM;                            // D
    float* bna    = bnsq  + DIM;                            // D
    float* bnb    = bna   + DIM;                            // D
    float* pooled = bnb   + DIM;                            // G*D
    float* counts = pooled + N_GRAPHS * DIM;                // G

    const int ND = N_NODES * DIM;

    // degree -> deg_isqrt (with self-loop)
    k_zero<<<(N_NODES + 255) / 256, 256, 0, stream>>>(dis, N_NODES);
    k_deg<<<(N_EDGES + 255) / 256, 256, 0, stream>>>(dst, dis, N_EDGES);
    k_disqrt<<<(N_NODES + 255) / 256, 256, 0, stream>>>(dis, N_NODES);

    const float* hcur = x;
    for (int l = 0; l < N_LAYERS; ++l) {
        k_gemm_bf16<<<1024, 256, 0, stream>>>(hcur, mask, W + (size_t)l * DIM * DIM,
                                              hwbuf, N_NODES);
        k_zero<<<(ND + 255) / 256, 256, 0, stream>>>(aggbuf, ND);
        k_zero<<<1, 256, 0, stream>>>(bnsum, 2 * DIM);   // bnsum + bnsq contiguous
        k_scatter<<<(N_EDGES * 32 + 255) / 256, 256, 0, stream>>>(src, dst, dis,
                                                                  hwbuf, aggbuf, N_EDGES);
        k_selfbias_stats<<<512, 128, 0, stream>>>(aggbuf, hwbuf, dis, b + l * DIM,
                                                  bnsum, bnsq, N_NODES);
        k_bn_finalize<<<1, 128, 0, stream>>>(bnsum, bnsq, gamma + l * DIM, beta + l * DIM,
                                             bna, bnb, N_NODES);
        k_bn_relu<<<(ND + 255) / 256, 256, 0, stream>>>(aggbuf, bna, bnb, hbuf, ND);
        hcur = hbuf;
    }

    k_zero<<<(N_GRAPHS * (DIM + 1) + 255) / 256, 256, 0, stream>>>(pooled,
                                                                   N_GRAPHS * (DIM + 1));
    k_pool<<<512, 128, 0, stream>>>(hbuf, mask, batch, pooled, N_NODES);
    k_counts<<<(N_NODES + 255) / 256, 256, 0, stream>>>(batch, counts, N_NODES);
    k_out<<<1, 64, 0, stream>>>(pooled, counts, Wout, bout, out);
}
